// RotaryAttention_77206332113018
// MI455X (gfx1250) — matrix operand order repaired and checked by reading, not compiled, not run on hardware
//
#include <hip/hip_runtime.h>
#include <hip/hip_bf16.h>
#include <math.h>

// ---------------------------------------------------------------------------
// RotaryAttention for MI455X (gfx1250): bf16 WMMA pipeline, f32 accumulate.
//   B=2, N=2048, QDIM=1024, HEADS=16, DIM_HEAD=64, INNER=1024
// ~69 GFLOP vs ~50MB traffic -> compute bound -> v_wmma_f32_16x16x32_bf16
// everywhere. GEMMs are 64x64 register-blocked per wave (16 wmma per 8KB of
// fragment loads, ~32 flops/byte from L2). __launch_bounds__(256,1) raises
// the VGPR budget so the 16 f32 accumulators stay resident (round-2 build
// spilled them to scratch under the default occupancy heuristic).
// Attention is computed transposed (sim^T = K*Q^T, out^T = V^T*P^T) so the
// online softmax is lane-local plus one xor-16 shuffle and P^T is rebuilt in
// registers (no LDS round trip); each wave carries two query tiles to reuse
// the K/V fragments.
// ---------------------------------------------------------------------------

typedef __attribute__((ext_vector_type(16))) __bf16 v16bf;
typedef __attribute__((ext_vector_type(8)))  __bf16 v8bf;
typedef __attribute__((ext_vector_type(8)))  float  v8f;

union BF16x16 { v16bf v; v8bf h[2]; };

constexpr int BATCH = 2;
constexpr int NSEQ  = 2048;
constexpr int QD    = 1024;
constexpr int HEADS = 16;
constexpr int DHEAD = 64;
constexpr int INNER = 1024;
constexpr int MROWS = BATCH * NSEQ;          // 4096
constexpr int PCOLS = 3 * INNER;             // q | k | v projection columns
constexpr float QK_SCALE = 0.125f;           // DIM_HEAD^-0.5

#define ZV8F {0.f,0.f,0.f,0.f,0.f,0.f,0.f,0.f}

static __device__ __forceinline__ v8f wmma_bf16(v16bf a, v16bf b, v8f c) {
  return __builtin_amdgcn_wmma_f32_16x16x32_bf16(false, a, false, b,
                                                 (short)0, c, false, false);
}

// A-fragment (16x32 bf16, row-major source, leading dim ld):
// lane<16 : row=l,    K = k0+{0..15}
// lane>=16: row=l-16, K = k0+{16..31}
static __device__ __forceinline__ v16bf
load_a_frag(const __hip_bfloat16* base, int ld, int row0, int k0) {
  const int l  = threadIdx.x & 31;
  const int m  = row0 + (l & 15);
  const int ko = k0 + ((l >> 4) << 4);
  const __hip_bfloat16* p = base + (size_t)m * ld + ko;
  BF16x16 r;
  r.h[0] = *(const v8bf*)(p);
  r.h[1] = *(const v8bf*)(p + 8);
  return r.v;
}

// B-fragment (32x16 bf16). Source stores matrix COLUMNS as rows (length ld):
// lane holds column n = col0+(l&15), K = k0 + (l>=16 ? 16 : 0) + 0..15
static __device__ __forceinline__ v16bf
load_b_frag(const __hip_bfloat16* base, int ld, int col0, int k0) {
  const int l  = threadIdx.x & 31;
  const int n  = col0 + (l & 15);
  const int ko = k0 + ((l >> 4) << 4);
  const __hip_bfloat16* p = base + (size_t)n * ld + ko;
  BF16x16 r;
  r.h[0] = *(const v8bf*)(p);
  r.h[1] = *(const v8bf*)(p + 8);
  return r.v;
}

// ---------------------------------------------------------------------------
// Pre-pass kernels (bandwidth-trivial): f32 -> bf16 convert / transpose.
// ---------------------------------------------------------------------------
__global__ void cvt_f32_bf16_kernel(const float* __restrict__ in,
                                    __hip_bfloat16* __restrict__ out, int n) {
  int i = blockIdx.x * blockDim.x + threadIdx.x;
  if (i < n) out[i] = __float2bfloat16(in[i]);
}

// dst[c*K + k] = src[k*C + c]
__global__ void transpose_cvt_kernel(const float* __restrict__ src,
                                     __hip_bfloat16* __restrict__ dst,
                                     int K, int C) {
  int idx = blockIdx.x * blockDim.x + threadIdx.x;
  if (idx < K * C) {
    int c = idx / K, k = idx - c * K;
    dst[idx] = __float2bfloat16(src[(size_t)k * C + c]);
  }
}

// ---------------------------------------------------------------------------
// Projection epilogue for one 16x16 C-tile (C layout: lane=col, vgpr+half=row)
// ---------------------------------------------------------------------------
static __device__ __forceinline__ void
store_proj_tile(v8f acc, int m0, int c0,
                __hip_bfloat16* __restrict__ qbuf,
                __hip_bfloat16* __restrict__ kbuf,
                __hip_bfloat16* __restrict__ vTbuf, int bs, int ro) {
  const int l   = threadIdx.x & 31;
  const int col = c0 + (l & 15);
  if (c0 < 2 * INNER) {                    // ---- q or k: apply RoPE ----
    const bool is_q = (c0 < INNER);
    const int inner = col & (INNER - 1);
    // inv_freq = 10000^(-(2*(inner/2))/INNER); ln(10000)=9.2103403720
    const float inv_freq =
        __expf(-(float)(inner & ~1) * (9.210340371976184f / (float)INNER));
    const float sgn = (col & 1) ? 1.0f : -1.0f;   // even: -sin, odd: +sin
    const float osc = is_q ? QK_SCALE : 1.0f;
    __hip_bfloat16* dst = is_q ? qbuf : kbuf;
    const int h = inner >> 6, d = inner & 63;
#pragma unroll
    for (int r = 0; r < 8; ++r) {
      const int m = m0 + r + ((l >> 4) << 3);
      const int n = m & (NSEQ - 1);
      const int b = m >> 11;
      float s, c;
      __sincosf((float)(n / bs + ro) * inv_freq, &s, &c);
      const float v  = acc[r];
      const float pv = __shfl_xor(v, 1);          // RoPE pair partner
      dst[(((size_t)b * HEADS + h) * NSEQ + n) * DHEAD + d] =
          __float2bfloat16((v * c + pv * s * sgn) * osc);
    }
  } else {                                  // ---- v: store transposed ----
    const int inner = col - 2 * INNER;
    const int h = inner >> 6, d = inner & 63;
#pragma unroll
    for (int r = 0; r < 8; ++r) {
      const int m = m0 + r + ((l >> 4) << 3);
      const int n = m & (NSEQ - 1);
      const int b = m >> 11;
      vTbuf[(((size_t)b * HEADS + h) * DHEAD + d) * NSEQ + n] =
          __float2bfloat16(acc[r]);
    }
  }
}

// ---------------------------------------------------------------------------
// Projection GEMM + RoPE: one wave computes a 64x64 block of
// [MROWS x PCOLS] = xb(4096x1024) @ wT(3072x1024, columns-contiguous).
// 16 resident f32 accumulators; 16 wmma per 32-wide K step.
// ---------------------------------------------------------------------------
__global__ void __launch_bounds__(256, 1)
proj_rope_kernel(const __hip_bfloat16* __restrict__ xb,
                 const __hip_bfloat16* __restrict__ wT,
                 __hip_bfloat16* __restrict__ qbuf,
                 __hip_bfloat16* __restrict__ kbuf,
                 __hip_bfloat16* __restrict__ vTbuf,
                 const int* __restrict__ bs_p,
                 const int* __restrict__ ro_p) {
  const int wave = blockIdx.x * (blockDim.x >> 5) + (threadIdx.x >> 5);
  const int CT = PCOLS / 64;                    // 48 column blocks
  const int mt = wave / CT, ct = wave % CT;
  const int m0 = mt * 64, c0 = ct * 64;

  v8f acc[4][4];
#pragma unroll
  for (int mi = 0; mi < 4; ++mi)
#pragma unroll
    for (int ci = 0; ci < 4; ++ci) acc[mi][ci] = (v8f)ZV8F;

  for (int k0 = 0; k0 < QD; k0 += 32) {
    v16bf a[4], b[4];
#pragma unroll
    for (int mi = 0; mi < 4; ++mi)
      a[mi] = load_a_frag(xb, QD, m0 + 16 * mi, k0);
#pragma unroll
    for (int ci = 0; ci < 4; ++ci)
      b[ci] = load_b_frag(wT, QD, c0 + 16 * ci, k0);
    if (k0 + 32 < QD) {                         // gfx1250 global_prefetch_b8
      __builtin_prefetch(xb + (size_t)m0 * QD + k0 + 32, 0, 3);
      __builtin_prefetch(wT + (size_t)c0 * QD + k0 + 32, 0, 3);
    }
#pragma unroll
    for (int mi = 0; mi < 4; ++mi)
#pragma unroll
      for (int ci = 0; ci < 4; ++ci)
        acc[mi][ci] = wmma_bf16(a[mi], b[ci], acc[mi][ci]);
  }

  const int bs = max(1, *bs_p);
  const int ro = *ro_p;
#pragma unroll
  for (int mi = 0; mi < 4; ++mi)
#pragma unroll
    for (int ci = 0; ci < 4; ++ci)
      store_proj_tile(acc[mi][ci], m0 + 16 * mi, c0 + 16 * ci,
                      qbuf, kbuf, vTbuf, bs, ro);
}

// ---------------------------------------------------------------------------
// Flash attention (transposed). Each wave: TWO 16-row query tiles sharing the
// K and V^T fragments of every 32-key step -> 16 wmma per 8 fragment loads.
// ---------------------------------------------------------------------------
__global__ void __launch_bounds__(256, 1)
attn_kernel(const __hip_bfloat16* __restrict__ qbuf,
            const __hip_bfloat16* __restrict__ kbuf,
            const __hip_bfloat16* __restrict__ vTbuf,
            __hip_bfloat16* __restrict__ attn_out) {
  const int bh = blockIdx.x;
  const int b = bh >> 4, h = bh & 15;
  const int wave = threadIdx.x >> 5;
  const int i0 = blockIdx.y * 256 + wave * 32;   // 32 query rows per wave
  const int l = threadIdx.x & 31;
  const bool lo = (l < 16);

  const __hip_bfloat16* qrow = qbuf + (((size_t)b * HEADS + h) * NSEQ) * DHEAD;
  const __hip_bfloat16* krow = kbuf + (((size_t)b * HEADS + h) * NSEQ) * DHEAD;
  const __hip_bfloat16* vT   = vTbuf + (((size_t)b * HEADS + h) * DHEAD) * NSEQ;

  v16bf qb[2][2];
#pragma unroll
  for (int it = 0; it < 2; ++it) {
    qb[it][0] = load_b_frag(qrow, DHEAD, i0 + 16 * it, 0);
    qb[it][1] = load_b_frag(qrow, DHEAD, i0 + 16 * it, 32);
  }

  v8f o[2][4];
  float mst[2], lst[2];
#pragma unroll
  for (int it = 0; it < 2; ++it) {
    mst[it] = -__builtin_inff();
    lst[it] = 0.0f;
#pragma unroll
    for (int db = 0; db < 4; ++db) o[it][db] = (v8f)ZV8F;
  }

  for (int j0 = 0; j0 < NSEQ; j0 += 32) {
    // shared fragments for this key block
    v16bf ka[4], va[4];
    ka[0] = load_a_frag(krow, DHEAD, j0, 0);
    ka[1] = load_a_frag(krow, DHEAD, j0, 32);
    ka[2] = load_a_frag(krow, DHEAD, j0 + 16, 0);
    ka[3] = load_a_frag(krow, DHEAD, j0 + 16, 32);
#pragma unroll
    for (int db = 0; db < 4; ++db)
      va[db] = load_a_frag(vT, NSEQ, db * 16, j0);

#pragma unroll
    for (int it = 0; it < 2; ++it) {
      // sim^T: rows j (softmax axis), cols i
      v8f s0 = (v8f)ZV8F, s1 = (v8f)ZV8F;
      s0 = wmma_bf16(ka[0], qb[it][0], s0);
      s0 = wmma_bf16(ka[1], qb[it][1], s0);
      s1 = wmma_bf16(ka[2], qb[it][0], s1);
      s1 = wmma_bf16(ka[3], qb[it][1], s1);

      // online softmax over j: 16 in-lane regs + one xor-16 shuffle
      float t = s0[0];
#pragma unroll
      for (int r = 0; r < 8; ++r) { t = fmaxf(t, s0[r]); t = fmaxf(t, s1[r]); }
      t = fmaxf(t, __shfl_xor(t, 16));
      const float mnew = fmaxf(mst[it], t);
      const float corr = __expf(mst[it] - mnew);
      float p0[8], p1[8], ssum = 0.0f;
#pragma unroll
      for (int r = 0; r < 8; ++r) {
        p0[r] = __expf(s0[r] - mnew);
        p1[r] = __expf(s1[r] - mnew);
        ssum += p0[r] + p1[r];
      }
      ssum += __shfl_xor(ssum, 16);
      lst[it] = lst[it] * corr + ssum;
      mst[it] = mnew;

      // rebuild P^T B-frag from C-layout tiles via xor-16 (no LDS)
      BF16x16 pb;
#pragma unroll
      for (int r = 0; r < 8; ++r) {
        const float x0 = p0[r], x1 = p1[r];
        const float y0 = __shfl_xor(x0, 16);
        const float y1 = __shfl_xor(x1, 16);
        pb.v[r]     = (__bf16)(lo ? x0 : y1);
        pb.v[r + 8] = (__bf16)(lo ? y0 : x1);
      }

      // out^T += V^T * P^T with rescaled accumulators
#pragma unroll
      for (int db = 0; db < 4; ++db) {
#pragma unroll
        for (int r = 0; r < 8; ++r) o[it][db][r] *= corr;
        o[it][db] = wmma_bf16(va[db], pb.v, o[it][db]);
      }
    }
  }

  // normalize + store: lane holds column i; rows d = db*16 + r + 8*(l>>4)
  const int dsub = (l >> 4) << 3;
#pragma unroll
  for (int it = 0; it < 2; ++it) {
    const float inv_l = 1.0f / lst[it];
    const int i = i0 + 16 * it + (l & 15);
    __hip_bfloat16* row =
        attn_out + ((size_t)b * NSEQ + i) * INNER + h * DHEAD;
#pragma unroll
    for (int db = 0; db < 4; ++db) {
      v8bf pack;
#pragma unroll
      for (int r = 0; r < 8; ++r) pack[r] = (__bf16)(o[it][db][r] * inv_l);
      *(v8bf*)(row + db * 16 + dsub) = pack;  // 8 consecutive d -> one b128
    }
  }
}

// ---------------------------------------------------------------------------
// Output GEMM (64x64 register-blocked): out = attn_bf16 @ w_out + b_out (f32)
// ---------------------------------------------------------------------------
__global__ void __launch_bounds__(256, 1)
out_gemm_kernel(const __hip_bfloat16* __restrict__ attn,
                const __hip_bfloat16* __restrict__ woT,
                const float* __restrict__ b_out,
                float* __restrict__ out) {
  const int wave = blockIdx.x * (blockDim.x >> 5) + (threadIdx.x >> 5);
  const int CT = QD / 64;                       // 16 column blocks
  const int mt = wave / CT, ct = wave % CT;
  const int m0 = mt * 64, c0 = ct * 64;

  v8f acc[4][4];
#pragma unroll
  for (int mi = 0; mi < 4; ++mi)
#pragma unroll
    for (int ci = 0; ci < 4; ++ci) acc[mi][ci] = (v8f)ZV8F;

  for (int k0 = 0; k0 < INNER; k0 += 32) {
    v16bf a[4], b[4];
#pragma unroll
    for (int mi = 0; mi < 4; ++mi)
      a[mi] = load_a_frag(attn, INNER, m0 + 16 * mi, k0);
#pragma unroll
    for (int ci = 0; ci < 4; ++ci)
      b[ci] = load_b_frag(woT, INNER, c0 + 16 * ci, k0);
    if (k0 + 32 < INNER) {
      __builtin_prefetch(attn + (size_t)m0 * INNER + k0 + 32, 0, 3);
      __builtin_prefetch(woT + (size_t)c0 * INNER + k0 + 32, 0, 3);
    }
#pragma unroll
    for (int mi = 0; mi < 4; ++mi)
#pragma unroll
      for (int ci = 0; ci < 4; ++ci)
        acc[mi][ci] = wmma_bf16(a[mi], b[ci], acc[mi][ci]);
  }

  const int l = threadIdx.x & 31;
#pragma unroll
  for (int ci = 0; ci < 4; ++ci) {
    const int c = c0 + 16 * ci + (l & 15);
    const float bias = b_out[c];
#pragma unroll
    for (int mi = 0; mi < 4; ++mi) {
#pragma unroll
      for (int r = 0; r < 8; ++r) {
        const int m = m0 + 16 * mi + r + ((l >> 4) << 3);
        out[(size_t)m * QD + c] = acc[mi][ci][r] + bias;
      }
    }
  }
}

// ---------------------------------------------------------------------------
// Host launcher.  Workspace usage: 48 MiB.
//   xb 8M | wT 6M | woT 2M | q 8M | k 8M | vT 8M | attn 8M
// ---------------------------------------------------------------------------
extern "C" void kernel_launch(void* const* d_in, const int* in_sizes, int n_in,
                              void* d_out, int out_size, void* d_ws,
                              size_t ws_size, hipStream_t stream) {
  const float* x     = (const float*)d_in[0];
  const float* w_q   = (const float*)d_in[1];
  const float* w_kv  = (const float*)d_in[2];
  const float* w_out = (const float*)d_in[3];
  const float* b_out = (const float*)d_in[4];
  const int* bs_p    = (const int*)d_in[5];
  const int* ro_p    = (const int*)d_in[6];
  float* out = (float*)d_out;

  char* w = (char*)d_ws;
  __hip_bfloat16* xb   = (__hip_bfloat16*)w; w += (size_t)MROWS * QD * 2;
  __hip_bfloat16* wT   = (__hip_bfloat16*)w; w += (size_t)PCOLS * QD * 2;
  __hip_bfloat16* woT  = (__hip_bfloat16*)w; w += (size_t)INNER * QD * 2;
  __hip_bfloat16* qbuf = (__hip_bfloat16*)w; w += (size_t)MROWS * INNER * 2;
  __hip_bfloat16* kbuf = (__hip_bfloat16*)w; w += (size_t)MROWS * INNER * 2;
  __hip_bfloat16* vTb  = (__hip_bfloat16*)w; w += (size_t)MROWS * INNER * 2;
  __hip_bfloat16* attn = (__hip_bfloat16*)w;

  { // 1) x -> bf16
    int n = MROWS * QD;
    cvt_f32_bf16_kernel<<<(n + 255) / 256, 256, 0, stream>>>(x, xb, n);
  }
  // 2) transpose+convert weights (columns contiguous for B-fragments)
  transpose_cvt_kernel<<<(QD * INNER + 255) / 256, 256, 0, stream>>>(
      w_q, wT, QD, INNER);
  transpose_cvt_kernel<<<(QD * 2 * INNER + 255) / 256, 256, 0, stream>>>(
      w_kv, wT + (size_t)INNER * QD, QD, 2 * INNER);
  transpose_cvt_kernel<<<(INNER * QD + 255) / 256, 256, 0, stream>>>(
      w_out, woT, INNER, QD);

  // 3) projections + RoPE: (4096/64)*(3072/64) = 3072 waves, 8 per block
  proj_rope_kernel<<<(MROWS / 64) * (PCOLS / 64) / 8, 256, 0, stream>>>(
      xb, wT, qbuf, kbuf, vTb, bs_p, ro_p);

  // 4) attention: grid (B*H, N/(8 waves * 32 rows)) = (32, 8)
  attn_kernel<<<dim3(BATCH * HEADS, NSEQ / 256), 256, 0, stream>>>(
      qbuf, kbuf, vTb, attn);

  // 5) output projection + bias: (4096/64)*(1024/64) = 1024 waves
  out_gemm_kernel<<<(MROWS / 64) * (QD / 64) / 8, 256, 0, stream>>>(
      attn, woT, b_out, out);
}
